// GNNPredictor_85822036509056
// MI455X (gfx1250) — matrix operand order repaired and checked by reading, not compiled, not run on hardware
//
#include <hip/hip_runtime.h>
#include <hip/hip_bf16.h>

// ---------------------------------------------------------------------------
// GNNPredictor for MI455X (gfx1250, wave32, WMMA).
// All GEMMs run through v_wmma_f32_16x16x32_bf16; BN affine + per-row scale
// are fused into the GEMM A-tile load. Staging uses vectorized
// global_load_b128 -> ds_store_b64 on the (dominant) full-tile path.
// Input pointer order assumption: m, node_feature, t1_feature, then params
// flattened in alphabetical key order (JAX pytree convention).
// ---------------------------------------------------------------------------

typedef __attribute__((ext_vector_type(16))) __bf16 v16bf;
typedef __attribute__((ext_vector_type(8)))  float  v8f;

#define BZ   256
#define ROI  360
#define EMB  8

enum { ACT_NONE = 0, ACT_LEAKY = 1, ACT_RELU = 2 };

__device__ __forceinline__ unsigned short f2bf(float f) {
    unsigned int u = __float_as_uint(f);
    u += 0x7FFFu + ((u >> 16) & 1u);      // round-to-nearest-even
    return (unsigned short)(u >> 16);
}
__device__ __forceinline__ unsigned int pk2bf(float lo, float hi) {
    return (unsigned int)f2bf(lo) | ((unsigned int)f2bf(hi) << 16);
}

// ---------------------------------------------------------------------------
// Fused GEMM:  C[M,N] = act( (colaff? A*cs+cb : A) * (rscale? rs[row] : 1) @ B + bias )
// A: MxK f32 row-major, B: KxN f32 row-major.
// 256 threads = 8 waves; block tile 128(M) x 32(N); wave tile 32x16
// (two v_wmma_f32_16x16x32_bf16 per K-step sharing one B fragment).
// REQUIREMENT: M % 128 == 0 (true for all call sites: 92160, 256).
// ---------------------------------------------------------------------------
#define AST 40   // padded LDS row stride in bf16 elems (80B -> conflict-free b128)

template<int ACT, bool RSCALE, bool COLAFF>
__global__ __launch_bounds__(256)
void gemm_bf16_kernel(const float* __restrict__ A, const float* __restrict__ B,
                      const float* __restrict__ bias,
                      const float* __restrict__ rs,
                      const float* __restrict__ cs, const float* __restrict__ cb,
                      float* __restrict__ C, int M, int N, int K)
{
    __shared__ __align__(16) unsigned short As[128 * AST];
    __shared__ __align__(16) unsigned short Bs[32 * AST];

    const int tid   = threadIdx.x;
    const int wave  = tid >> 5;
    const int lane  = tid & 31;
    const int halfW = lane >> 4;       // 0 | 1
    const int l15   = lane & 15;
    const int wy    = wave >> 1;       // 0..3  (M sub-tile)
    const int wx    = wave & 1;        // 0..1  (N sub-tile)
    const int mBase = blockIdx.y * 128;
    const int nBase = blockIdx.x * 32;

    // staging coords (invariant over K loop)
    const int aC4   = tid & 7;         // float4 column within 32-wide K slab
    const int aR0   = tid >> 3;        // base row, +32 per unroll step
    const int bK    = tid >> 3;        // B row (k) within slab
    const int bN4   = tid & 7;         // B float4 column within 32-wide N slab

    // fragment coords
    const int aRow0 = wy * 32 + l15;
    const int aRow1 = aRow0 + 16;
    const int aOff  = halfW * 8;       // K offset of low 8 elems (ISA A layout)
    const int bRow  = wx * 16 + l15;
    const int bOff  = halfW * 8;       // K offset of low 8 elems (same as A layout)

    const bool nFull = (nBase + 32 <= N) && ((N & 3) == 0);

    // preload per-row scales once (rows handled by this thread are fixed)
    float rsv[4] = {1.f, 1.f, 1.f, 1.f};
    if (RSCALE) {
        #pragma unroll
        for (int i = 0; i < 4; ++i) rsv[i] = rs[mBase + aR0 + i * 32];
    }

    v8f acc0 = {};
    v8f acc1 = {};

    for (int k0 = 0; k0 < K; k0 += 32) {
        const bool kFull = (k0 + 32 <= K);

        // ---------------- stage A tile: 128 x 32 ----------------
        if (kFull) {
            float4 csv, cbv;
            if (COLAFF) {
                csv = *(const float4*)&cs[k0 + aC4 * 4];
                cbv = *(const float4*)&cb[k0 + aC4 * 4];
            }
            #pragma unroll
            for (int i = 0; i < 4; ++i) {
                int row = aR0 + i * 32;
                float4 v = *(const float4*)&A[(size_t)(mBase + row) * K + k0 + aC4 * 4];
                if (COLAFF) {
                    v.x = v.x * csv.x + cbv.x;  v.y = v.y * csv.y + cbv.y;
                    v.z = v.z * csv.z + cbv.z;  v.w = v.w * csv.w + cbv.w;
                }
                if (RSCALE) { v.x *= rsv[i]; v.y *= rsv[i]; v.z *= rsv[i]; v.w *= rsv[i]; }
                uint2 p; p.x = pk2bf(v.x, v.y); p.y = pk2bf(v.z, v.w);
                *(uint2*)&As[row * AST + aC4 * 4] = p;
            }
        } else {
            #pragma unroll
            for (int i = 0; i < 4; ++i) {
                int row = aR0 + i * 32;
                float e[4];
                #pragma unroll
                for (int j = 0; j < 4; ++j) {
                    int gk = k0 + aC4 * 4 + j;
                    float v = 0.f;
                    if (gk < K) {
                        v = A[(size_t)(mBase + row) * K + gk];
                        if (COLAFF) v = v * cs[gk] + cb[gk];
                        if (RSCALE) v *= rsv[i];
                    }
                    e[j] = v;
                }
                uint2 p; p.x = pk2bf(e[0], e[1]); p.y = pk2bf(e[2], e[3]);
                *(uint2*)&As[row * AST + aC4 * 4] = p;
            }
        }

        // ---------------- stage B tile transposed: Bt[n][k], 32 x 32 ----------------
        if (kFull && nFull) {
            float4 v = *(const float4*)&B[(size_t)(k0 + bK) * N + nBase + bN4 * 4];
            Bs[(bN4 * 4 + 0) * AST + bK] = f2bf(v.x);
            Bs[(bN4 * 4 + 1) * AST + bK] = f2bf(v.y);
            Bs[(bN4 * 4 + 2) * AST + bK] = f2bf(v.z);
            Bs[(bN4 * 4 + 3) * AST + bK] = f2bf(v.w);
        } else {
            int gk = k0 + bK;
            #pragma unroll
            for (int j = 0; j < 4; ++j) {
                int gn = nBase + bN4 * 4 + j;
                float v = (gk < K && gn < N) ? B[(size_t)gk * N + gn] : 0.f;
                Bs[(bN4 * 4 + j) * AST + bK] = f2bf(v);
            }
        }
        __syncthreads();

        // ---------------- fragments per CDNA5 16-bit WMMA VGPR layouts ----------------
        union { v16bf v; uint4 q[2]; } fa0, fa1, fb;
        fa0.q[0] = *(const uint4*)&As[aRow0 * AST + aOff];
        fa0.q[1] = *(const uint4*)&As[aRow0 * AST + aOff + 16];
        fa1.q[0] = *(const uint4*)&As[aRow1 * AST + aOff];
        fa1.q[1] = *(const uint4*)&As[aRow1 * AST + aOff + 16];
        const uint4* pb = (const uint4*)&Bs[bRow * AST + bOff];
        fb.q[0] = pb[0];
        fb.q[1] = pb[2];

        acc0 = __builtin_amdgcn_wmma_f32_16x16x32_bf16(
                   false, fa0.v, false, fb.v, (short)0, acc0, false, false);
        acc1 = __builtin_amdgcn_wmma_f32_16x16x32_bf16(
                   false, fa1.v, false, fb.v, (short)0, acc1, false, false);
        __syncthreads();
    }

    // ---------------- epilogue: bias + activation ----------------
    int col = nBase + wx * 16 + l15;
    if (col < N) {
        float bv = bias ? bias[col] : 0.f;
        #pragma unroll
        for (int i = 0; i < 8; ++i) {
            int r0 = mBase + wy * 32 + halfW * 8 + i;
            float v0 = acc0[i] + bv;
            float v1 = acc1[i] + bv;
            if (ACT == ACT_LEAKY) { v0 = v0 > 0.f ? v0 : 0.2f * v0; v1 = v1 > 0.f ? v1 : 0.2f * v1; }
            if (ACT == ACT_RELU)  { v0 = v0 > 0.f ? v0 : 0.f;       v1 = v1 > 0.f ? v1 : 0.f; }
            C[(size_t)r0 * N + col]        = v0;
            C[(size_t)(r0 + 16) * N + col] = v1;
        }
    }
}

// ---------------------------------------------------------------------------
// s[row] = sum_k m[row, k]   (wave per row, shuffle reduce)
// ---------------------------------------------------------------------------
__global__ __launch_bounds__(256)
void rowsum_kernel(const float* __restrict__ m, float* __restrict__ s,
                   int rows, int cols)
{
    int wave = threadIdx.x >> 5, lane = threadIdx.x & 31;
    int row = blockIdx.x * 8 + wave;
    if (row >= rows) return;
    const float* p = m + (size_t)row * cols;
    float acc = 0.f;
    for (int c = lane; c < cols; c += 32) acc += p[c];
    for (int off = 16; off; off >>= 1) acc += __shfl_xor(acc, off, 32);
    if (lane == 0) s[row] = acc;
}

// ---------------------------------------------------------------------------
// BatchNorm1d (N,C) training stats -> folded affine cs = inv*g, cb = b-mu*inv*g
// ---------------------------------------------------------------------------
__global__ __launch_bounds__(256)
void bn_stats_kernel(const float* __restrict__ X, const float* __restrict__ g,
                     const float* __restrict__ b, float* __restrict__ cs,
                     float* __restrict__ cb, int rows, int C)
{
    __shared__ float sS[256], sQ[256];
    int tid = threadIdx.x;
    int c = blockIdx.x * 8 + (tid & 7);
    float s = 0.f, q = 0.f;
    if (c < C) {
        for (int r = tid >> 3; r < rows; r += 32) {
            float v = X[(size_t)r * C + c];
            s += v; q += v * v;
        }
    }
    sS[tid] = s; sQ[tid] = q;
    __syncthreads();
    for (int off = 128; off >= 8; off >>= 1) {
        if (tid < off) { sS[tid] += sS[tid + off]; sQ[tid] += sQ[tid + off]; }
        __syncthreads();
    }
    if (tid < 8 && c < C) {
        float mu  = sS[tid] / (float)rows;
        float var = sQ[tid] / (float)rows - mu * mu;
        float inv = rsqrtf(var + 1e-5f);
        float gg  = g[c];
        cs[c] = inv * gg;
        cb[c] = b[c] - mu * inv * gg;
    }
}

// ---------------------------------------------------------------------------
// bn3d stats: X viewed (BZ, ROI, EMB); per-roi-channel stats over (b, e).
// ---------------------------------------------------------------------------
__global__ __launch_bounds__(256)
void bn3_stats_kernel(const float* __restrict__ X, const float* __restrict__ g,
                      const float* __restrict__ b, float* __restrict__ cs,
                      float* __restrict__ cb)
{
    __shared__ float sS[256], sQ[256];
    int r = blockIdx.x, t = threadIdx.x;
    const float* p = X + (size_t)t * (ROI * EMB) + r * EMB;
    float s = 0.f, q = 0.f;
    #pragma unroll
    for (int e = 0; e < EMB; ++e) { float v = p[e]; s += v; q += v * v; }
    sS[t] = s; sQ[t] = q;
    __syncthreads();
    for (int off = 128; off; off >>= 1) {
        if (t < off) { sS[t] += sS[t + off]; sQ[t] += sQ[t + off]; }
        __syncthreads();
    }
    if (t == 0) {
        float n   = (float)(BZ * EMB);
        float mu  = sS[0] / n;
        float var = sQ[0] / n - mu * mu;
        float inv = rsqrtf(var + 1e-5f);
        float gg  = g[r];
        cs[r] = inv * gg;
        cb[r] = b[r] - mu * inv * gg;
    }
}

// ---------------------------------------------------------------------------
// attRaw[row] = softmax_k(Q[row]·K[b,k]) @ V[b]  -- two-pass softmax, one wave/row.
// ---------------------------------------------------------------------------
__global__ __launch_bounds__(256)
void attn_kernel(const float* __restrict__ Q, const float* __restrict__ Km,
                 const float* __restrict__ V, float* __restrict__ attRaw)
{
    __shared__ float sQ[8][32];
    int wave = threadIdx.x >> 5, lane = threadIdx.x & 31;
    int row = blockIdx.x * 8 + wave;            // [0, BZ*ROI)
    int b = row / ROI;
    sQ[wave][lane] = Q[(size_t)row * 32 + lane];
    __syncthreads();

    float sc[12];
    const float* Kb = Km + (size_t)b * ROI * 32;
    #pragma unroll
    for (int j = 0; j < 12; ++j) {
        int k = j * 32 + lane;
        if (k < ROI) {
            const float* kp = Kb + (size_t)k * 32;
            float d = 0.f;
            #pragma unroll
            for (int t = 0; t < 32; ++t) d += sQ[wave][t] * kp[t];
            sc[j] = d;
        } else {
            sc[j] = -3.0e38f;
        }
    }
    float mx = sc[0];
    #pragma unroll
    for (int j = 1; j < 12; ++j) mx = fmaxf(mx, sc[j]);
    for (int off = 16; off; off >>= 1) mx = fmaxf(mx, __shfl_xor(mx, off, 32));

    float se = 0.f, sv = 0.f;
    const float* Vb = V + (size_t)b * ROI;
    #pragma unroll
    for (int j = 0; j < 12; ++j) {
        int k = j * 32 + lane;
        if (k < ROI) {
            float e = __expf(sc[j] - mx);
            se += e; sv += e * Vb[k];
        }
    }
    for (int off = 16; off; off >>= 1) {
        se += __shfl_xor(se, off, 32);
        sv += __shfl_xor(sv, off, 32);
    }
    if (lane == 0) attRaw[row] = sv / se;
}

// ---------------------------------------------------------------------------
// attOut[b,r] = softmax_r(attRaw[b,:]) * ROI     (one block per batch)
// ---------------------------------------------------------------------------
__global__ __launch_bounds__(512)
void att_softmax_kernel(const float* __restrict__ attRaw, float* __restrict__ attOut)
{
    __shared__ float red[512];
    int b = blockIdx.x, t = threadIdx.x;
    float v = (t < ROI) ? attRaw[(size_t)b * ROI + t] : -3.0e38f;
    red[t] = v; __syncthreads();
    for (int off = 256; off; off >>= 1) {
        if (t < off) red[t] = fmaxf(red[t], red[t + off]);
        __syncthreads();
    }
    float mx = red[0];
    __syncthreads();
    float e = (t < ROI) ? __expf(v - mx) : 0.f;
    red[t] = e; __syncthreads();
    for (int off = 256; off; off >>= 1) {
        if (t < off) red[t] += red[t + off];
        __syncthreads();
    }
    float sum = red[0];
    if (t < ROI) attOut[(size_t)b * ROI + t] = e / sum * (float)ROI;
}

// ---------------------------------------------------------------------------
// xf = (bn3-affine(Xd)) * att, written straight into d_out's xf region.
// ---------------------------------------------------------------------------
__global__ __launch_bounds__(256)
void xf_kernel(const float* __restrict__ Xd, const float* __restrict__ cs,
               const float* __restrict__ cb, const float* __restrict__ att,
               float* __restrict__ xf)
{
    int idx = blockIdx.x * 256 + threadIdx.x;     // < BZ*ROI*EMB
    int b = idx / (ROI * EMB);
    int rem = idx - b * (ROI * EMB);
    int r = rem >> 3;
    float v = Xd[idx] * cs[r] + cb[r];
    xf[idx] = v * att[b * ROI + r];
}

// ---------------------------------------------------------------------------
// Workspace layout (floats). Q/K/V/attRaw alias XB after it dies.
// ---------------------------------------------------------------------------
static const size_t OFF_S   = 0;                         // 92160
static const size_t OFF_CS1 = 92160,  OFF_CB1 = 92520;
static const size_t OFF_CS2 = 92880,  OFF_CB2 = 93240;
static const size_t OFF_CS3 = 93600,  OFF_CB3 = 93960;
static const size_t OFF_H1  = 94464;                     // 256x256
static const size_t OFF_H2  = 160000;                    // 256x32
static const size_t OFF_G1  = 168448;                    // 256x128
static const size_t OFF_G2  = 201216;                    // 256x32
static const size_t OFF_XD  = 212992;                    // 92160x8
static const size_t OFF_XC  = 950272;                    // 92160x64
static const size_t OFF_XA  = 6848512;                   // 92160x360
static const size_t OFF_XB  = 40026112;                  // 92160x360
static const size_t OFF_Q   = OFF_XB;                    // 92160x32
static const size_t OFF_K2  = OFF_XB + 2949120;          // 92160x32
static const size_t OFF_V   = OFF_XB + 5898240;          // 92160
static const size_t OFF_AR  = OFF_XB + 5990400;          // 92160

// d_out offsets (floats): out_cls, out_site, xf, att, t1_red
static const size_t OUT_CLS = 0, OUT_SITE = 512, OUT_XF = 5632,
                    OUT_ATT = 742912, OUT_T1 = 835072;

extern "C" void kernel_launch(void* const* d_in, const int* in_sizes, int n_in,
                              void* d_out, int out_size, void* d_ws, size_t ws_size,
                              hipStream_t stream)
{
    (void)in_sizes; (void)n_in; (void)out_size; (void)ws_size;

    const float* m    = (const float*)d_in[0];
    const float* nf   = (const float*)d_in[1];
    const float* t1   = (const float*)d_in[2];
    // params, alphabetical (JAX pytree order):
    const float* bn1_b = (const float*)d_in[3];
    const float* bn1_g = (const float*)d_in[4];
    const float* bn2_b = (const float*)d_in[5];
    const float* bn2_g = (const float*)d_in[6];
    const float* bn3_b = (const float*)d_in[7];
    const float* bn3_g = (const float*)d_in[8];
    const float* fcn_b1 = (const float*)d_in[9];
    const float* fcn_b2 = (const float*)d_in[10];
    const float* fcn_b3 = (const float*)d_in[11];
    const float* fcn_w1 = (const float*)d_in[12];
    const float* fcn_w2 = (const float*)d_in[13];
    const float* fcn_w3 = (const float*)d_in[14];
    const float* gcn1_b = (const float*)d_in[15];
    const float* gcn1_w = (const float*)d_in[16];
    const float* gcn2_b1 = (const float*)d_in[17];
    const float* gcn2_b2 = (const float*)d_in[18];
    const float* gcn2_w1 = (const float*)d_in[19];
    const float* gcn2_w2 = (const float*)d_in[20];
    const float* gcn_b1 = (const float*)d_in[21];
    const float* gcn_b2 = (const float*)d_in[22];
    const float* gcn_w1 = (const float*)d_in[23];
    const float* gcn_w2 = (const float*)d_in[24];
    const float* k_b = (const float*)d_in[25];
    const float* k_w = (const float*)d_in[26];
    const float* q_b = (const float*)d_in[27];
    const float* q_w = (const float*)d_in[28];
    const float* sd_b1 = (const float*)d_in[29];
    const float* sd_b2 = (const float*)d_in[30];
    const float* sd_b3 = (const float*)d_in[31];
    const float* sd_w1 = (const float*)d_in[32];
    const float* sd_w2 = (const float*)d_in[33];
    const float* sd_w3 = (const float*)d_in[34];
    const float* t1_b = (const float*)d_in[35];
    const float* t1_w = (const float*)d_in[36];
    const float* v_b = (const float*)d_in[37];
    const float* v_w = (const float*)d_in[38];

    float* ws  = (float*)d_ws;
    float* out = (float*)d_out;

    float* S   = ws + OFF_S;
    float* CS1 = ws + OFF_CS1; float* CB1 = ws + OFF_CB1;
    float* CS2 = ws + OFF_CS2; float* CB2 = ws + OFF_CB2;
    float* CS3 = ws + OFF_CS3; float* CB3 = ws + OFF_CB3;
    float* XA  = ws + OFF_XA;  float* XB  = ws + OFF_XB;
    float* XC  = ws + OFF_XC;  float* XD  = ws + OFF_XD;
    float* Qb  = ws + OFF_Q;   float* Kb  = ws + OFF_K2;
    float* Vb  = ws + OFF_V;   float* AR  = ws + OFF_AR;
    float* H1  = ws + OFF_H1;  float* H2  = ws + OFF_H2;
    float* G1  = ws + OFF_G1;  float* G2  = ws + OFF_G2;

    const int MR = BZ * ROI;   // 92160

    #define GEMM(ACT, RS, CA, A_, B_, BIAS_, RSP_, CSP_, CBP_, C_, M_, N_, K_)      \
        gemm_bf16_kernel<ACT, RS, CA>                                               \
            <<<dim3(((N_) + 31) / 32, (M_) / 128), 256, 0, stream>>>                \
            (A_, B_, BIAS_, RSP_, CSP_, CBP_, C_, M_, N_, K_)

    // s = rowsum(m)
    rowsum_kernel<<<MR / 8, 256, 0, stream>>>(m, S, MR, ROI);

    // GCN trunk (BN affine + row-scale fused into A loads)
    GEMM(ACT_LEAKY, true,  false, nf, gcn_w1, gcn_b1, S, nullptr, nullptr, XA, MR, ROI, ROI);
    GEMM(ACT_NONE,  false, false, XA, gcn_w2, gcn_b2, nullptr, nullptr, nullptr, XB, MR, ROI, ROI);
    bn_stats_kernel<<<(ROI + 7) / 8, 256, 0, stream>>>(XB, bn1_g, bn1_b, CS1, CB1, MR, ROI);
    GEMM(ACT_LEAKY, true,  true,  XB, gcn1_w, gcn1_b, S, CS1, CB1, XA, MR, ROI, ROI);
    bn_stats_kernel<<<(ROI + 7) / 8, 256, 0, stream>>>(XA, bn2_g, bn2_b, CS2, CB2, MR, ROI);
    GEMM(ACT_LEAKY, true,  true,  XA, gcn2_w1, gcn2_b1, S, CS2, CB2, XC, MR, 64, ROI);
    GEMM(ACT_LEAKY, false, false, XC, gcn2_w2, gcn2_b2, nullptr, nullptr, nullptr, XD, MR, EMB, 64);
    bn3_stats_kernel<<<ROI, 256, 0, stream>>>(XD, bn3_g, bn3_b, CS3, CB3);

    // attention (XB is dead: Q/K/V/attRaw alias it)
    GEMM(ACT_NONE, false, false, m, q_w, q_b, nullptr, nullptr, nullptr, Qb, MR, 32, ROI);
    GEMM(ACT_NONE, false, false, m, k_w, k_b, nullptr, nullptr, nullptr, Kb, MR, 32, ROI);
    GEMM(ACT_NONE, false, false, m, v_w, v_b, nullptr, nullptr, nullptr, Vb, MR, 1, ROI);
    attn_kernel<<<MR / 8, 256, 0, stream>>>(Qb, Kb, Vb, AR);
    att_softmax_kernel<<<BZ, 512, 0, stream>>>(AR, out + OUT_ATT);

    // xf = bn3(x) * att  -> d_out xf region (also feeds the heads)
    float* xf = out + OUT_XF;
    xf_kernel<<<(BZ * ROI * EMB) / 256, 256, 0, stream>>>(XD, CS3, CB3, out + OUT_ATT, xf);

    // classification head
    GEMM(ACT_LEAKY, false, false, xf, fcn_w1, fcn_b1, nullptr, nullptr, nullptr, H1, BZ, 256, ROI * EMB);
    GEMM(ACT_LEAKY, false, false, H1, fcn_w2, fcn_b2, nullptr, nullptr, nullptr, H2, BZ, 32, 256);
    GEMM(ACT_NONE,  false, false, H2, fcn_w3, fcn_b3, nullptr, nullptr, nullptr, out + OUT_CLS, BZ, 2, 32);

    // site head (gradient reversal == identity forward)
    GEMM(ACT_LEAKY, false, false, xf, sd_w1, sd_b1, nullptr, nullptr, nullptr, G1, BZ, 128, ROI * EMB);
    GEMM(ACT_LEAKY, false, false, G1, sd_w2, sd_b2, nullptr, nullptr, nullptr, G2, BZ, 32, 128);
    GEMM(ACT_NONE,  false, false, G2, sd_w3, sd_b3, nullptr, nullptr, nullptr, out + OUT_SITE, BZ, 20, 32);

    // t1 reduction
    GEMM(ACT_RELU, false, false, t1, t1_w, t1_b, nullptr, nullptr, nullptr, out + OUT_T1, BZ, 928, 2416);

    #undef GEMM
}
